// Attention_52201032515992
// MI455X (gfx1250) — compile-verified, matrix operand order checked by reading, not run on hardware
//
#include <hip/hip_runtime.h>
#include <hip/hip_bf16.h>

#define B_    2
#define S_    2048
#define HID_  2048
#define HEADS_ 16
#define KVH_  4
#define HD_   128
#define KVDIM_ 512
#define MTOT_ (B_ * S_)          // 4096 rows
#define SCALE_ 0.08838834764831845f  // 1/sqrt(128)

typedef __attribute__((ext_vector_type(4)))  __bf16 v4bf;
typedef __attribute__((ext_vector_type(8)))  __bf16 v8bf;
typedef __attribute__((ext_vector_type(16))) __bf16 v16bf;
typedef __attribute__((ext_vector_type(8)))  float  v8f;
typedef __attribute__((ext_vector_type(4)))  int    v4i;

__device__ __forceinline__ v8bf ld8g(const __bf16* p) { return *(const v8bf*)p; }

__device__ __forceinline__ v16bf cat16(v8bf lo, v8bf hi) {
  v16bf r;
#pragma unroll
  for (int i = 0; i < 8; ++i) { r[i] = lo[i]; r[8 + i] = hi[i]; }
  return r;
}

__device__ __forceinline__ v8f wmma_bf16(v16bf a, v16bf b, v8f c) {
  return __builtin_amdgcn_wmma_f32_16x16x32_bf16(
      /*neg_a=*/false, a, /*neg_b=*/false, b,
      /*c_mod=*/(short)0, c, /*reuse_a=*/false, /*reuse_b=*/false);
}

__device__ __forceinline__ v8f v8f_zero() {
  v8f z;
#pragma unroll
  for (int i = 0; i < 8; ++i) z[i] = 0.0f;
  return z;
}

// ---------------------------------------------------------------------------
// CDNA5 async global -> LDS copy (16B per lane, tracked by ASYNCcnt).
// Builtin signature (probe-confirmed by round-2 diagnostic):
//   (AS1 int4* src, AS3 int4* dst, imm int offset, imm int cpol)
// Falls back to a synchronous VGPR round-trip if the builtin is unavailable.
// ---------------------------------------------------------------------------
#if defined(__has_builtin)
#if __has_builtin(__builtin_amdgcn_global_load_async_to_lds_b128)
#define HAVE_ASYNC_LDS 1
#endif
#endif

#ifdef HAVE_ASYNC_LDS
__device__ __forceinline__ void async_cp16(void* lds, const void* g) {
  __builtin_amdgcn_global_load_async_to_lds_b128(
      (__attribute__((address_space(1))) v4i*)(g),
      (__attribute__((address_space(3))) v4i*)(lds), /*offset=*/0, /*cpol=*/0);
}
__device__ __forceinline__ void async_wait0() {
  asm volatile("s_wait_asynccnt 0x0" ::: "memory");
}
#else
__device__ __forceinline__ void async_cp16(void* lds, const void* g) {
  *(v8bf*)lds = *(const v8bf*)g;
}
__device__ __forceinline__ void async_wait0() {
  asm volatile("s_wait_loadcnt 0x0" ::: "memory");
  asm volatile("s_wait_dscnt 0x0" ::: "memory");
}
#endif

// ---------------------------------------------------------------------------
// fp32 -> bf16 elementwise conversion (vectorized x4)
// ---------------------------------------------------------------------------
__global__ void cvt_f32_bf16_kernel(const float* __restrict__ in,
                                    __bf16* __restrict__ out, int n) {
  int i = (blockIdx.x * blockDim.x + threadIdx.x) * 4;
  if (i < n) {
    float4 f = *(const float4*)(in + i);
    v4bf o;
    o[0] = (__bf16)f.x; o[1] = (__bf16)f.y; o[2] = (__bf16)f.z; o[3] = (__bf16)f.w;
    *(v4bf*)(out + i) = o;
  }
}

// ---------------------------------------------------------------------------
// C[M,N] (fp32) = A[M,K] (bf16) * W[N,K]^T (bf16)
// Block: 256 threads (8 waves). Tile: 256 (M) x 128 (N), K-step 32,
// double-buffered LDS filled by async global->LDS copies.
// Wave (wm = w>>1, wn = w&1) computes 64x64 via 4x4 WMMA accumulators
// (16 WMMA : 16 ds_load_b128 per K-step).
// ---------------------------------------------------------------------------
#define LDS_STRIDE 40   // 32 + 8 pad (bf16 elems), keeps 16B alignment

__global__ __launch_bounds__(256)
void gemm_bf16_kernel(const __bf16* __restrict__ A,
                      const __bf16* __restrict__ W,
                      float* __restrict__ C,
                      int M, int N, int K) {
  __shared__ __bf16 As[2][256][LDS_STRIDE];
  __shared__ __bf16 Bs[2][128][LDS_STRIDE];

  const int lane = threadIdx.x & 31;
  const int wave = threadIdx.x >> 5;
  const int wm = wave >> 1;          // 0..3
  const int wn = wave & 1;           // 0..1
  const int m0 = blockIdx.y * 256;
  const int n0 = blockIdx.x * 128;
  const int laneN = lane & 15;
  const int kboff = (lane >> 4) * 8; // 0 or 8 (K-halves per ISA A/B layout)
  const int rowoff = (lane >> 4) * 8;

  auto stage = [&](int k0, int buf) {
#pragma unroll
    for (int it = 0; it < 4; ++it) {           // A: 256 rows x 4 chunks of 8
      int j = threadIdx.x + it * 256;
      int row = j >> 2, col = (j & 3) * 8;
      async_cp16(&As[buf][row][col], A + (size_t)(m0 + row) * K + k0 + col);
    }
#pragma unroll
    for (int it = 0; it < 2; ++it) {           // B: 128 rows x 4 chunks of 8
      int j = threadIdx.x + it * 256;
      int row = j >> 2, col = (j & 3) * 8;
      async_cp16(&Bs[buf][row][col], W + (size_t)(n0 + row) * K + k0 + col);
    }
  };

  v8f acc[4][4];
#pragma unroll
  for (int mi = 0; mi < 4; ++mi)
#pragma unroll
    for (int ni = 0; ni < 4; ++ni) acc[mi][ni] = v8f_zero();

  stage(0, 0);
  for (int k0 = 0; k0 < K; k0 += 32) {
    const int buf = (k0 >> 5) & 1;
    async_wait0();
    __syncthreads();
    if (k0 + 32 < K) stage(k0 + 32, buf ^ 1);

    v16bf afrag[4];
#pragma unroll
    for (int mi = 0; mi < 4; ++mi) {
      int r = wm * 64 + mi * 16 + laneN;
      afrag[mi] = cat16(*(const v8bf*)&As[buf][r][kboff],
                        *(const v8bf*)&As[buf][r][kboff + 16]);
    }
    v16bf bfrag[4];
#pragma unroll
    for (int ni = 0; ni < 4; ++ni) {
      int r = wn * 64 + ni * 16 + laneN;
      bfrag[ni] = cat16(*(const v8bf*)&Bs[buf][r][kboff],
                        *(const v8bf*)&Bs[buf][r][kboff + 16]);
    }

#pragma unroll
    for (int mi = 0; mi < 4; ++mi)
#pragma unroll
      for (int ni = 0; ni < 4; ++ni)
        acc[mi][ni] = wmma_bf16(afrag[mi], bfrag[ni], acc[mi][ni]);
  }

  // store: D layout row = r + 8*(lane>>4), col = lane&15
#pragma unroll
  for (int mi = 0; mi < 4; ++mi)
#pragma unroll
    for (int ni = 0; ni < 4; ++ni) {
      int rbase = m0 + wm * 64 + mi * 16 + rowoff;
      int col = n0 + wn * 64 + ni * 16 + laneN;
#pragma unroll
      for (int r = 0; r < 8; ++r)
        C[(size_t)(rbase + r) * N + col] = acc[mi][ni][r];
    }
}

// ---------------------------------------------------------------------------
// RoPE + layout change: in fp32 [B,S,nh*HD] -> out bf16 [B,nh,S,HD]
// ---------------------------------------------------------------------------
__global__ void rope_to_bf16_kernel(const float* __restrict__ in,
                                    const float* __restrict__ cosT,
                                    const float* __restrict__ sinT,
                                    __bf16* __restrict__ out, int nh) {
  int i = blockIdx.x * blockDim.x + threadIdx.x;  // pair index
  int total = B_ * S_ * nh * (HD_ / 2);
  if (i >= total) return;
  int p = i % (HD_ / 2);
  int h = (i / (HD_ / 2)) % nh;
  int s = (i / (HD_ / 2 * nh)) % S_;
  int b = i / (HD_ / 2 * nh * S_);
  size_t src = ((size_t)(b * S_ + s) * nh + h) * HD_ + 2 * p;
  float ev = in[src], od = in[src + 1];
  float c = cosT[s * (HD_ / 2) + p], sn = sinT[s * (HD_ / 2) + p];
  size_t dst = ((size_t)(b * nh + h) * S_ + s) * HD_ + 2 * p;
  out[dst]     = (__bf16)(ev * c - od * sn);
  out[dst + 1] = (__bf16)(ev * sn + od * c);
}

// V: fp32 [B,S,KVDIM] -> bf16 transposed [B,KVH,HD,S]
__global__ void v_transpose_kernel(const float* __restrict__ in,
                                   __bf16* __restrict__ out) {
  int i = blockIdx.x * blockDim.x + threadIdx.x;
  int total = B_ * S_ * KVDIM_;
  if (i >= total) return;
  int d = i % HD_;
  int h = (i / HD_) % KVH_;
  int s = (i / KVDIM_) % S_;
  int b = i / (KVDIM_ * S_);
  out[((size_t)(b * KVH_ + h) * HD_ + d) * S_ + s] =
      (__bf16)in[(size_t)(b * S_ + s) * KVDIM_ + h * HD_ + d];
}

// ---------------------------------------------------------------------------
// Flash attention (causal, GQA). Block: 256 thr = 8 waves; block owns 128
// query rows (16 per wave). K/V tiles for each 32-key block are staged
// cooperatively into double-buffered LDS with async copies; waves outside
// their causal range skip compute but keep barrier participation.
// Q [B,H,S,HD] bf16, K [B,KVH,S,HD] bf16, Vt [B,KVH,HD,S] bf16.
// Output y fp32 [B,S,HID] in (s, h*HD+d) layout.
// ---------------------------------------------------------------------------
#define FA_PAD 8

__global__ __launch_bounds__(256)
void flash_attn_kernel(const __bf16* __restrict__ Q,
                       const __bf16* __restrict__ Kt,
                       const __bf16* __restrict__ Vt,
                       float* __restrict__ Y) {
  __shared__ __bf16 Ks[2][32][HD_ + FA_PAD];   // key-major:  [key][d]
  __shared__ __bf16 Vs[2][HD_][32 + FA_PAD];   // d-major:    [d][key]
  __shared__ __bf16 Pls[8][16][32];            // wave-private P staging

  const int lane = threadIdx.x & 31;
  const int w = threadIdx.x >> 5;
  const int b = blockIdx.z;
  const int h = blockIdx.y;
  const int qb0 = blockIdx.x * 128;
  const int q0 = qb0 + w * 16;
  const int kvh = h / (HEADS_ / KVH_);
  const int laneN = lane & 15;
  const int kboff = (lane >> 4) * 8;
  const int rowoff = (lane >> 4) * 8;

  const __bf16* Qp = Q + ((size_t)(b * HEADS_ + h) * S_ + q0) * HD_;
  const __bf16* Kp = Kt + (size_t)(b * KVH_ + kvh) * S_ * HD_;
  const __bf16* Vp = Vt + (size_t)(b * KVH_ + kvh) * HD_ * (size_t)S_;

  auto stage = [&](int kb, int buf) {
#pragma unroll
    for (int it = 0; it < 2; ++it) {           // K: 32 rows x 16 chunks of 8
      int j = threadIdx.x + it * 256;
      int row = j >> 4, col = (j & 15) * 8;
      async_cp16(&Ks[buf][row][col], Kp + (size_t)(kb + row) * HD_ + col);
    }
#pragma unroll
    for (int it = 0; it < 2; ++it) {           // V: 128 rows x 4 chunks of 8
      int j = threadIdx.x + it * 256;
      int row = j >> 2, col = (j & 3) * 8;
      async_cp16(&Vs[buf][row][col], Vp + (size_t)row * S_ + kb + col);
    }
  };

  // Q fragments for the 4 K-steps over HD=128 (loaded once)
  v16bf qfr[4];
#pragma unroll
  for (int ks = 0; ks < 4; ++ks) {
    const __bf16* p = Qp + (size_t)laneN * HD_ + ks * 32 + kboff;
    qfr[ks] = cat16(ld8g(p), ld8g(p + 16));
  }

  v8f o[8];
#pragma unroll
  for (int ni = 0; ni < 8; ++ni) o[ni] = v8f_zero();
  float mrow[8], lrow[8];
#pragma unroll
  for (int r = 0; r < 8; ++r) { mrow[r] = -3.0e38f; lrow[r] = 0.0f; }

  const int qmax_w = q0 + 15;
  const int kb_end = qb0 + 128;    // causal limit for the whole block
  stage(0, 0);
  for (int kb = 0; kb < kb_end; kb += 32) {
    const int buf = (kb >> 5) & 1;
    async_wait0();
    __syncthreads();
    if (kb + 32 < kb_end) stage(kb + 32, buf ^ 1);

    if (kb <= qmax_w) {
      float sc[2][8];
#pragma unroll
      for (int t = 0; t < 2; ++t) {
        v8f s = v8f_zero();
#pragma unroll
        for (int ks = 0; ks < 4; ++ks) {
          v16bf kf = cat16(*(const v8bf*)&Ks[buf][t * 16 + laneN][ks * 32 + kboff],
                           *(const v8bf*)&Ks[buf][t * 16 + laneN][ks * 32 + kboff + 16]);
          s = wmma_bf16(qfr[ks], kf, s);
        }
        int key = kb + t * 16 + laneN;
#pragma unroll
        for (int r = 0; r < 8; ++r) {
          int row = q0 + r + rowoff;
          sc[t][r] = (key <= row) ? s[r] * SCALE_ : -3.0e38f;
        }
      }

      // online softmax per row (16-lane group reductions)
#pragma unroll
      for (int r = 0; r < 8; ++r) {
        float m = fmaxf(sc[0][r], sc[1][r]);
        m = fmaxf(m, __shfl_xor(m, 8, 16));
        m = fmaxf(m, __shfl_xor(m, 4, 16));
        m = fmaxf(m, __shfl_xor(m, 2, 16));
        m = fmaxf(m, __shfl_xor(m, 1, 16));
        float mnew = fmaxf(mrow[r], m);
        float p0 = __expf(sc[0][r] - mnew);
        float p1 = __expf(sc[1][r] - mnew);
        sc[0][r] = p0; sc[1][r] = p1;
        float lsum = p0 + p1;
        lsum += __shfl_xor(lsum, 8, 16);
        lsum += __shfl_xor(lsum, 4, 16);
        lsum += __shfl_xor(lsum, 2, 16);
        lsum += __shfl_xor(lsum, 1, 16);
        float f = __expf(mrow[r] - mnew);
        lrow[r] = lrow[r] * f + lsum;
        mrow[r] = mnew;
#pragma unroll
        for (int ni = 0; ni < 8; ++ni) o[ni][r] *= f;
      }

      // P (D layout) -> LDS -> A-fragment layout (wave-private, dscnt wait)
#pragma unroll
      for (int t = 0; t < 2; ++t)
#pragma unroll
        for (int r = 0; r < 8; ++r)
          Pls[w][r + rowoff][t * 16 + laneN] = (__bf16)sc[t][r];
      asm volatile("s_wait_dscnt 0x0" ::: "memory");

      v16bf pfr = cat16(*(const v8bf*)&Pls[w][laneN][kboff],
                        *(const v8bf*)&Pls[w][laneN][kboff + 16]);

#pragma unroll
      for (int ni = 0; ni < 8; ++ni) {
        v16bf vf = cat16(*(const v8bf*)&Vs[buf][ni * 16 + laneN][kboff],
                         *(const v8bf*)&Vs[buf][ni * 16 + laneN][kboff + 16]);
        o[ni] = wmma_bf16(pfr, vf, o[ni]);
      }
    }
  }

  // epilogue: normalize and scatter to y [B,S,HID]
#pragma unroll
  for (int ni = 0; ni < 8; ++ni)
#pragma unroll
    for (int r = 0; r < 8; ++r) {
      int row = q0 + r + rowoff;
      Y[((size_t)b * S_ + row) * HID_ + h * HD_ + ni * 16 + laneN] =
          o[ni][r] / lrow[r];
    }
}

// ---------------------------------------------------------------------------
// Launcher
// ---------------------------------------------------------------------------
extern "C" void kernel_launch(void* const* d_in, const int* in_sizes, int n_in,
                              void* d_out, int out_size, void* d_ws, size_t ws_size,
                              hipStream_t stream) {
  const float* x    = (const float*)d_in[0];
  const float* cosT = (const float*)d_in[1];
  const float* sinT = (const float*)d_in[2];
  const float* Wq   = (const float*)d_in[3];
  const float* Wk   = (const float*)d_in[4];
  const float* Wv   = (const float*)d_in[5];
  const float* Wo   = (const float*)d_in[6];
  float* out = (float*)d_out;

  char* ws = (char*)d_ws;
  size_t off = 0;
  __bf16* xb  = (__bf16*)(ws + off); off += (size_t)MTOT_ * HID_ * 2;      // 16 MB
  __bf16* wqb = (__bf16*)(ws + off); off += (size_t)HID_ * HID_ * 2;       //  8 MB
  __bf16* wkb = (__bf16*)(ws + off); off += (size_t)KVDIM_ * HID_ * 2;     //  2 MB
  __bf16* wvb = (__bf16*)(ws + off); off += (size_t)KVDIM_ * HID_ * 2;     //  2 MB
  __bf16* wob = (__bf16*)(ws + off); off += (size_t)HID_ * HID_ * 2;       //  8 MB
  float*  qf  = (float*)(ws + off);  off += (size_t)MTOT_ * HID_ * 4;      // 32 MB (reused as yf)
  float*  kf  = (float*)(ws + off);  off += (size_t)MTOT_ * KVDIM_ * 4;    //  8 MB (reused as yb)
  float*  vf  = (float*)(ws + off);  off += (size_t)MTOT_ * KVDIM_ * 4;    //  8 MB
  __bf16* qb2 = (__bf16*)(ws + off); off += (size_t)MTOT_ * HID_ * 2;      // 16 MB
  __bf16* kb2 = (__bf16*)(ws + off); off += (size_t)MTOT_ * KVDIM_ * 2;    //  4 MB
  __bf16* vtb = (__bf16*)(ws + off); off += (size_t)MTOT_ * KVDIM_ * 2;    //  4 MB
  float*  yf = qf;                       // alias: qf dead after rope
  __bf16* yb = (__bf16*)kf;              // alias: kf/vf dead after rope/transpose

  const int T = 256;
  // 1. fp32 -> bf16 conversions
  { int n = MTOT_ * HID_;   cvt_f32_bf16_kernel<<<n / 4 / T, T, 0, stream>>>(x,  xb,  n); }
  { int n = HID_ * HID_;    cvt_f32_bf16_kernel<<<n / 4 / T, T, 0, stream>>>(Wq, wqb, n); }
  { int n = KVDIM_ * HID_;  cvt_f32_bf16_kernel<<<n / 4 / T, T, 0, stream>>>(Wk, wkb, n); }
  { int n = KVDIM_ * HID_;  cvt_f32_bf16_kernel<<<n / 4 / T, T, 0, stream>>>(Wv, wvb, n); }
  { int n = HID_ * HID_;    cvt_f32_bf16_kernel<<<n / 4 / T, T, 0, stream>>>(Wo, wob, n); }

  // 2. QKV projections
  gemm_bf16_kernel<<<dim3(HID_ / 128, MTOT_ / 256), T, 0, stream>>>(xb, wqb, qf, MTOT_, HID_, HID_);
  gemm_bf16_kernel<<<dim3(KVDIM_ / 128, MTOT_ / 256), T, 0, stream>>>(xb, wkb, kf, MTOT_, KVDIM_, HID_);
  gemm_bf16_kernel<<<dim3(KVDIM_ / 128, MTOT_ / 256), T, 0, stream>>>(xb, wvb, vf, MTOT_, KVDIM_, HID_);

  // 3. RoPE + layout to bf16
  { int n = B_ * S_ * HEADS_ * (HD_ / 2);
    rope_to_bf16_kernel<<<(n + T - 1) / T, T, 0, stream>>>(qf, cosT, sinT, qb2, HEADS_); }
  { int n = B_ * S_ * KVH_ * (HD_ / 2);
    rope_to_bf16_kernel<<<(n + T - 1) / T, T, 0, stream>>>(kf, cosT, sinT, kb2, KVH_); }
  { int n = B_ * S_ * KVDIM_;
    v_transpose_kernel<<<(n + T - 1) / T, T, 0, stream>>>(vf, vtb); }

  // 4. causal flash attention
  flash_attn_kernel<<<dim3(S_ / 128, HEADS_, B_), T, 0, stream>>>(qb2, kb2, vtb, yf);

  // 5. output projection
  { int n = MTOT_ * HID_;   cvt_f32_bf16_kernel<<<n / 4 / T, T, 0, stream>>>(yf, yb, n); }
  gemm_bf16_kernel<<<dim3(HID_ / 128, MTOT_ / 256), T, 0, stream>>>(yb, wob, out, MTOT_, HID_, HID_);
}
